// RBFN_16896401342544
// MI455X (gfx1250) — matrix operand order repaired and checked by reading, not compiled, not run on hardware
//
#include <hip/hip_runtime.h>
#include <hip/hip_bf16.h>

// ---------------------------------------------------------------------------
// RBFN forward, fused, for MI455X (gfx1250, wave32, WMMA).
//   B = X@Z  (16384 x 4096, K=512)  via v_wmma_f32_16x16x32_f16
//   pred = alpha + exp(-gamma*(|x|^2+|z|^2-2B)) @ beta
// Never materializes B/A. f16 inputs, f32 accumulation.
// Round 2: double-buffered LDS tile fed by global_load_async_to_lds_b128
// (ASYNCcnt path), double-buffered B fragments to relax dscnt waits.
// ---------------------------------------------------------------------------

typedef __attribute__((ext_vector_type(16))) _Float16 v16h;
typedef __attribute__((ext_vector_type(8)))  _Float16 v8h;
typedef __attribute__((ext_vector_type(8)))  float    v8f;

#define BATCH  16384
#define NIN    512
#define NCENT  4096

#define MTILE  64     // rows per block (4 row-tiles of 16)
#define NSTEP  32     // n columns staged per iteration (2 n-groups of 16)
#define NITER  (NCENT / NSTEP)

union ABfrag { v16h v; v8h h[2]; };

struct alignas(8)  H4 { _Float16 h[4]; };
struct alignas(16) H8 { _Float16 h[8]; };

// ---------------------------------------------------------------------------
// Stage X: fp32 [16384][512] -> f16 row-major + per-row |x|^2.
// ---------------------------------------------------------------------------
__global__ __launch_bounds__(256) void stage_x(const float* __restrict__ X,
                                               _Float16* __restrict__ Xh,
                                               float* __restrict__ xsq) {
    const int wave = (blockIdx.x * blockDim.x + threadIdx.x) >> 5;
    const int lane = threadIdx.x & 31;
    if (wave >= BATCH) return;

    const float4* xr = (const float4*)(X + (size_t)wave * NIN);
    H4* xo = (H4*)(Xh + (size_t)wave * NIN);

    float s = 0.0f;
#pragma unroll
    for (int j = 0; j < 4; ++j) {           // 4 * 32 lanes * float4 = 512
        const float4 v = xr[j * 32 + lane];
        s += v.x * v.x + v.y * v.y + v.z * v.z + v.w * v.w;
        H4 o;
        o.h[0] = (_Float16)v.x; o.h[1] = (_Float16)v.y;
        o.h[2] = (_Float16)v.z; o.h[3] = (_Float16)v.w;
        xo[j * 32 + lane] = o;
    }
#pragma unroll
    for (int m = 1; m < 32; m <<= 1) s += __shfl_xor(s, m, 32);
    if (lane == 0) xsq[wave] = s;
}

// ---------------------------------------------------------------------------
// Stage Z: fp32 [512][4096] -> f16 TRANSPOSED Zt[4096][512] + per-col |z|^2.
// ---------------------------------------------------------------------------
__global__ __launch_bounds__(256) void stage_z(const float* __restrict__ Z,
                                               _Float16* __restrict__ Zt,
                                               float* __restrict__ zsq) {
    const int c = blockIdx.x * blockDim.x + threadIdx.x;   // 0..4095
    if (c >= NCENT) return;

    _Float16* zo = Zt + (size_t)c * NIN;
    float s = 0.0f;
    for (int k0 = 0; k0 < NIN; k0 += 8) {
        H8 o;
#pragma unroll
        for (int j = 0; j < 8; ++j) {
            const float v = Z[(size_t)(k0 + j) * NCENT + c];  // coalesced
            s += v * v;
            o.h[j] = (_Float16)v;
        }
        *(H8*)(zo + k0) = o;
    }
    zsq[c] = s;
}

// ---------------------------------------------------------------------------
// Async global -> LDS copy of one 32 KB Zt tile (32 rows x 512 f16).
// 256 threads x 8 x b128 = 32 KB. Tracked by ASYNCcnt; caller waits with
// s_wait_asynccnt 0 before the consuming barrier.
// ---------------------------------------------------------------------------
__device__ __forceinline__ void async_copy_tile(const _Float16* __restrict__ g,
                                                _Float16* l, int tid) {
    unsigned long long ga = (unsigned long long)(uintptr_t)g + (unsigned)tid * 16u;
    // generic LDS pointer: low 32 bits are the LDS byte offset
    unsigned la = (unsigned)(uintptr_t)l + (unsigned)tid * 16u;
#pragma unroll
    for (int i = 0; i < 8; ++i) {
        asm volatile("global_load_async_to_lds_b128 %0, %1, off"
                     :: "v"(la + (unsigned)i * 4096u),
                        "v"(ga + (unsigned long long)i * 4096ull)
                     : "memory");
    }
}

__device__ __forceinline__ void wait_async0() {
    asm volatile("s_wait_asynccnt 0x0" ::: "memory");
}

// ---------------------------------------------------------------------------
// Main fused kernel.
// Block: 256 threads = 8 waves. Waves 0-3 / 4-7 split N; wave&3 picks row tile.
// A fragments (16 rows x K=512) live in 128 VGPRs for the whole kernel.
// Zt tiles double-buffered in LDS, prefetched asynchronously.
// ---------------------------------------------------------------------------
__global__ __launch_bounds__(256, 1) void rbfn_main(
        const _Float16* __restrict__ Xh, const _Float16* __restrict__ Zt,
        const float* __restrict__ xsq,  const float* __restrict__ zsq,
        const float* __restrict__ beta, const float* __restrict__ alpha,
        const float* __restrict__ log_gamma, float* __restrict__ out) {

    __shared__ __align__(16) _Float16 zTile[2][NSTEP * NIN];   // 2 x 32 KB
    __shared__ float partial[8][16];

    const int tid    = threadIdx.x;
    const int wave   = tid >> 5;
    const int lane   = tid & 31;
    const int l15    = lane & 15;
    const int hi     = lane >> 4;          // half-wave select
    const int rowTil = wave & 3;           // 4 row tiles of 16
    const int nGrp   = wave >> 2;          // 2 n-groups of 16
    const int rowBase = blockIdx.x * MTILE + rowTil * 16;

    const float gamma = __expf(log_gamma[0]);

    // --- Preload A fragments in WMMA lane layout -------------------------
    // lane l (row = rowBase + (l&15)) holds K in {kb8..kb8+7} U {kb8+16..kb8+23}
    // per 32-wide k-tile, kb8 = (l>=16) ? 8 : 0  ->  two contiguous v8h loads.
    ABfrag aF[16];
    const _Float16* xr = Xh + (size_t)(rowBase + l15) * NIN + hi * 8;
#pragma unroll
    for (int kb = 0; kb < 16; ++kb) {
        aF[kb].h[0] = *(const v8h*)(xr + kb * 32);
        aF[kb].h[1] = *(const v8h*)(xr + kb * 32 + 16);
    }

    // |x|^2 for the 8 accumulator rows this lane sees (C layout: M = r + 8*hi)
    float xs[8];
#pragma unroll
    for (int r = 0; r < 8; ++r) xs[r] = xsq[rowBase + hi * 8 + r];

    float racc[8];
#pragma unroll
    for (int r = 0; r < 8; ++r) racc[r] = 0.0f;

    // prologue: prefetch first tile
    async_copy_tile(Zt, zTile[0], tid);

    for (int it = 0; it < NITER; ++it) {
        const int n0  = it * NSTEP;
        const int cur = it & 1;

        wait_async0();            // my async writes into zTile[cur] landed
        __syncthreads();          // everyone's landed; prev reads of [cur^1] done

        if (it + 1 < NITER)       // overlap next prefetch with this compute
            async_copy_tile(Zt + (size_t)(n0 + NSTEP) * NIN, zTile[cur ^ 1], tid);

        // B fragment: lane l = column (nGrp*16 + l&15), K = {kb8..kb8+7} U {kb8+16..kb8+23}
        const int nl = nGrp * 16 + l15;
        const _Float16* zr = zTile[cur] + nl * NIN + hi * 8;

        const int   n  = n0 + nl;
        const float zq = zsq[n];
        const float bt = beta[n];

        v8f acc = {};
        ABfrag bF[2];
        bF[0].h[0] = *(const v8h*)(zr);
        bF[0].h[1] = *(const v8h*)(zr + 16);
#pragma unroll
        for (int kb = 0; kb < 16; ++kb) {
            if (kb + 1 < 16) {    // double-buffer next B fragment
                bF[(kb + 1) & 1].h[0] = *(const v8h*)(zr + (kb + 1) * 32);
                bF[(kb + 1) & 1].h[1] = *(const v8h*)(zr + (kb + 1) * 32 + 16);
            }
            acc = __builtin_amdgcn_wmma_f32_16x16x32_f16(
                /*neg_a=*/false, aF[kb].v, /*neg_b=*/false, bF[kb & 1].v,
                /*c_mod=*/(short)0, acc, /*reuse_a=*/false, /*reuse_b=*/false);
        }

        // Fused epilogue: D = |x|^2 + |z|^2 - 2B ; A = exp(-gamma*D) ; += A*beta
#pragma unroll
        for (int r = 0; r < 8; ++r) {
            const float d = xs[r] + zq - 2.0f * acc[r];
            racc[r] += __expf(-gamma * d) * bt;
        }
    }

    // --- Deterministic reduction over the 16 lanes of each half-wave -----
#pragma unroll
    for (int m = 1; m < 16; m <<= 1) {
#pragma unroll
        for (int r = 0; r < 8; ++r) racc[r] += __shfl_xor(racc[r], m, 32);
    }
    if (l15 == 0) {
#pragma unroll
        for (int r = 0; r < 8; ++r) partial[wave][hi * 8 + r] = racc[r];
    }
    __syncthreads();

    // combine the two n-group waves of each row tile; one store per row
    if (tid < MTILE) {
        const int rt = tid >> 4;
        const int rw = tid & 15;
        const float s = partial[rt][rw] + partial[rt + 4][rw];
        out[blockIdx.x * MTILE + tid] = alpha[0] + s;
    }
}

// ---------------------------------------------------------------------------
extern "C" void kernel_launch(void* const* d_in, const int* in_sizes, int n_in,
                              void* d_out, int out_size, void* d_ws, size_t ws_size,
                              hipStream_t stream) {
    const float* X         = (const float*)d_in[0];   // [16384,512]
    const float* alpha     = (const float*)d_in[1];   // [1]
    const float* log_gamma = (const float*)d_in[2];   // [1]
    const float* beta      = (const float*)d_in[3];   // [4096]
    const float* Z         = (const float*)d_in[4];   // [512,4096]
    float* out = (float*)d_out;                       // [16384]

    char* ws = (char*)d_ws;
    const size_t XH_BYTES = (size_t)BATCH * NIN * sizeof(_Float16);   // 16 MiB
    const size_t ZT_BYTES = (size_t)NCENT * NIN * sizeof(_Float16);   //  4 MiB
    _Float16* Xh  = (_Float16*)ws;
    _Float16* Zt  = (_Float16*)(ws + XH_BYTES);
    float*    xsq = (float*)(ws + XH_BYTES + ZT_BYTES);
    float*    zsq = xsq + BATCH;

    stage_x<<<(BATCH * 32) / 256, 256, 0, stream>>>(X, Xh, xsq);
    stage_z<<<NCENT / 256, 256, 0, stream>>>(Z, Zt, zsq);
    rbfn_main<<<BATCH / MTILE, 256, 0, stream>>>(Xh, Zt, xsq, zsq, beta,
                                                 alpha, log_gamma, out);
}
